// GaussianUpsampling_64914135712017
// MI455X (gfx1250) — matrix operand order repaired and checked by reading, not compiled, not run on hardware
//
#include <hip/hip_runtime.h>
#include <hip/hip_bf16.h>

typedef __attribute__((ext_vector_type(16))) _Float16 v16h;
typedef __attribute__((ext_vector_type(8)))  _Float16 v8h;
typedef __attribute__((ext_vector_type(8)))  float    v8f;

// ---------------------------------------------------------------------------
// Kernel 1: centers c[b,n] = cumsum(dur)[b,n] - 0.5*dur[b,n]
// One block per batch, Hillis-Steele inclusive scan over N (<=512) in LDS.
// ---------------------------------------------------------------------------
__global__ __launch_bounds__(512)
void gup_centers_kernel(const float* __restrict__ dur,
                        float* __restrict__ centers, int N) {
  __shared__ float s[512];
  const int b = blockIdx.x;
  const int t = threadIdx.x;
  float v = (t < N) ? dur[(size_t)b * N + t] : 0.0f;
  s[t] = v;
  __syncthreads();
  for (int off = 1; off < 512; off <<= 1) {
    float add = (t >= off) ? s[t - off] : 0.0f;
    __syncthreads();
    s[t] += add;
    __syncthreads();
  }
  if (t < N) centers[(size_t)b * N + t] = s[t] - 0.5f * v;
}

// ---------------------------------------------------------------------------
// Kernel 2: inT[b,h,n] = (f16) inputs[b,n,h]  (transpose + downconvert)
// Puts the WMMA B-operand K-dimension (n) contiguous in memory.
// ---------------------------------------------------------------------------
__global__ __launch_bounds__(256)
void gup_cvtT_kernel(const float* __restrict__ in, _Float16* __restrict__ inT,
                     int B, int N, int H) {
  size_t i = (size_t)blockIdx.x * 256 + threadIdx.x;
  size_t total = (size_t)B * N * H;
  if (i >= total) return;
  int h = (int)(i % H);
  size_t bn = i / H;
  int n = (int)(bn % N);
  int b = (int)(bn / N);
  inT[((size_t)b * H + h) * N + n] = (_Float16)in[i];
}

// ---------------------------------------------------------------------------
// Kernel 3: fused masked-softmax + WMMA contraction.
// Block = (b, 64-row t-tile). 512 threads = 16 waves. Wave w owns h-tiles
// {w, w+16} (H == 512 -> 32 h-tiles); 4 M-tiles of 16 rows each, so every
// B-fragment feeds 4 WMMAs. Assumes N <= 512, N % 32 == 0, H == 512.
// ---------------------------------------------------------------------------
#define TM 64                       // t-rows per block
#define NSTRIDE 520                 // 512 + 8 halves: row stride = 260 dwords
                                    // == 4 (mod 64 banks) -> conflict-free ds_b128

__global__ __launch_bounds__(512)
void gup_main_kernel(const _Float16* __restrict__ inT,
                     const float* __restrict__ centers,
                     const long long* __restrict__ inputs_len,
                     float* __restrict__ out,
                     int B, int N, int H, int T) {
  __shared__ _Float16 w16[TM][NSTRIDE];  // softmax weights, row = t-row, col = n
  __shared__ float    c_lds[512];
  __shared__ float    red[TM][8];

  const int b   = blockIdx.y;
  const int t0  = blockIdx.x * TM;
  const int tid = threadIdx.x;
  const int lane = tid & 31;
  const int wave = tid >> 5;

  for (int n = tid; n < N; n += 512)
    c_lds[n] = centers[(size_t)b * N + n];
  __syncthreads();

  // ---- masked softmax over n: 8 sub-workers per t-row, 64 rows ----
  const int len   = (int)inputs_len[b];
  const int r     = tid >> 3;        // t-row 0..63
  const int sub   = tid & 7;
  const int chunk = N >> 3;
  const int nbeg  = sub * chunk;
  const int nend  = nbeg + chunk;
  const float tval = (float)(t0 + r);

  float m = -3.0e38f;
  for (int n = nbeg; n < nend; ++n) {
    if (n < len) {
      float d = tval - c_lds[n];
      m = fmaxf(m, -0.5f * d * d);
    }
  }
  red[r][sub] = m;
  __syncthreads();
  float rm = red[r][0];
  #pragma unroll
  for (int j = 1; j < 8; ++j) rm = fmaxf(rm, red[r][j]);
  __syncthreads();

  float sum = 0.0f;
  for (int n = nbeg; n < nend; ++n) {
    if (n < len) {
      float d = tval - c_lds[n];
      sum += __expf(-0.5f * d * d - rm);
    }
  }
  red[r][sub] = sum;
  __syncthreads();
  float rs = 0.0f;
  #pragma unroll
  for (int j = 0; j < 8; ++j) rs += red[r][j];
  const float inv = 1.0f / rs;

  for (int n = nbeg; n < nend; ++n) {
    float w = 0.0f;
    if (n < len) {
      float d = tval - c_lds[n];
      w = __expf(-0.5f * d * d - rm) * inv;
    }
    w16[r][n] = (_Float16)w;
  }
  __syncthreads();

  // ---- WMMA: out[t,h] = sum_n W[t,n] * In[n,h]; 4 M-tiles x 2 H-tiles ----
  const int mrow   = lane & 15;
  const int laneHi = lane >> 4;
  const int offA   = laneHi * 8;     // 16-bit A layout: K {0..7,16..23}/{8..15,24..31}
  const int offB   = laneHi * 8;     // 16-bit B layout: K {0..7,16..23}/{8..15,24..31}

  const _Float16* arow0 = &w16[mrow][0];
  const _Float16* arow1 = &w16[16 + mrow][0];
  const _Float16* arow2 = &w16[32 + mrow][0];
  const _Float16* arow3 = &w16[48 + mrow][0];

  const size_t bH = (size_t)b * H;
  const int h0 = wave * 16;                       // h-tiles: h0 and h0+256
  const _Float16* bp0 = inT + (bH + h0 +   0 + mrow) * (size_t)N;
  const _Float16* bp1 = inT + (bH + h0 + 256 + mrow) * (size_t)N;

  v8f acc[4][2];
  #pragma unroll
  for (int i = 0; i < 4; ++i)
    #pragma unroll
    for (int j = 0; j < 2; ++j)
      acc[i][j] = (v8f){};

  for (int n0 = 0; n0 < N; n0 += 32) {
    // A fragments from LDS (conflict-free thanks to NSTRIDE padding)
    v8h l0 = *(const v8h*)(arow0 + n0 + offA);
    v8h h0v = *(const v8h*)(arow0 + n0 + offA + 16);
    v16h a0 = __builtin_shufflevector(l0, h0v,
        0,1,2,3,4,5,6,7,8,9,10,11,12,13,14,15);
    l0  = *(const v8h*)(arow1 + n0 + offA);
    h0v = *(const v8h*)(arow1 + n0 + offA + 16);
    v16h a1 = __builtin_shufflevector(l0, h0v,
        0,1,2,3,4,5,6,7,8,9,10,11,12,13,14,15);
    l0  = *(const v8h*)(arow2 + n0 + offA);
    h0v = *(const v8h*)(arow2 + n0 + offA + 16);
    v16h a2 = __builtin_shufflevector(l0, h0v,
        0,1,2,3,4,5,6,7,8,9,10,11,12,13,14,15);
    l0  = *(const v8h*)(arow3 + n0 + offA);
    h0v = *(const v8h*)(arow3 + n0 + offA + 16);
    v16h a3 = __builtin_shufflevector(l0, h0v,
        0,1,2,3,4,5,6,7,8,9,10,11,12,13,14,15);

    // B fragment 0 (global, L2-resident), reused by 4 M-tiles
    v8h q0 = *(const v8h*)(bp0 + n0 + offB);
    v8h q1 = *(const v8h*)(bp0 + n0 + offB + 16);
    v16h bm = __builtin_shufflevector(q0, q1,
        0,1,2,3,4,5,6,7,8,9,10,11,12,13,14,15);
    acc[0][0] = __builtin_amdgcn_wmma_f32_16x16x32_f16(false, a0, false, bm,
                                                       (short)0, acc[0][0], false, false);
    acc[1][0] = __builtin_amdgcn_wmma_f32_16x16x32_f16(false, a1, false, bm,
                                                       (short)0, acc[1][0], false, false);
    acc[2][0] = __builtin_amdgcn_wmma_f32_16x16x32_f16(false, a2, false, bm,
                                                       (short)0, acc[2][0], false, false);
    acc[3][0] = __builtin_amdgcn_wmma_f32_16x16x32_f16(false, a3, false, bm,
                                                       (short)0, acc[3][0], false, false);

    // B fragment 1
    q0 = *(const v8h*)(bp1 + n0 + offB);
    q1 = *(const v8h*)(bp1 + n0 + offB + 16);
    bm = __builtin_shufflevector(q0, q1,
        0,1,2,3,4,5,6,7,8,9,10,11,12,13,14,15);
    acc[0][1] = __builtin_amdgcn_wmma_f32_16x16x32_f16(false, a0, false, bm,
                                                       (short)0, acc[0][1], false, false);
    acc[1][1] = __builtin_amdgcn_wmma_f32_16x16x32_f16(false, a1, false, bm,
                                                       (short)0, acc[1][1], false, false);
    acc[2][1] = __builtin_amdgcn_wmma_f32_16x16x32_f16(false, a2, false, bm,
                                                       (short)0, acc[2][1], false, false);
    acc[3][1] = __builtin_amdgcn_wmma_f32_16x16x32_f16(false, a3, false, bm,
                                                       (short)0, acc[3][1], false, false);
  }

  // ---- store C/D fragments (f32 16x16 layout: VGPR rr -> M=rr / M=rr+8) ----
  #pragma unroll
  for (int mt = 0; mt < 4; ++mt) {
    #pragma unroll
    for (int rr = 0; rr < 8; ++rr) {
      const int mo = mt * 16 + (laneHi ? (rr + 8) : rr);
      float* orow = out + ((size_t)b * T + t0 + mo) * H + mrow;
      orow[h0 +   0] = acc[mt][0][rr];
      orow[h0 + 256] = acc[mt][1][rr];
    }
  }
}

// ---------------------------------------------------------------------------
extern "C" void kernel_launch(void* const* d_in, const int* in_sizes, int n_in,
                              void* d_out, int out_size, void* d_ws, size_t ws_size,
                              hipStream_t stream) {
  const float*     inputs     = (const float*)d_in[0];
  const long long* inputs_len = (const long long*)d_in[1];  // int64 per reference
  const float*     durations  = (const float*)d_in[2];

  const int B  = in_sizes[1];             // inputs_len: [B]
  const int BN = in_sizes[2];             // durations:  [B,N]
  const int N  = BN / B;
  const int H  = in_sizes[0] / BN;        // inputs: [B,N,H]
  const int T  = out_size / (B * H);      // out:    [B,T,H]

  // workspace: centers (B*N f32) | inT (B*H*N f16)
  float* centers = (float*)d_ws;
  size_t coff = (((size_t)B * N * sizeof(float)) + 255) & ~(size_t)255;
  _Float16* inT = (_Float16*)((char*)d_ws + coff);

  gup_centers_kernel<<<B, 512, 0, stream>>>(durations, centers, N);

  size_t total = (size_t)B * N * H;
  gup_cvtT_kernel<<<(unsigned)((total + 255) / 256), 256, 0, stream>>>(
      inputs, inT, B, N, H);

  dim3 grid(T / TM, B);
  gup_main_kernel<<<grid, 512, 0, stream>>>(
      inT, centers, inputs_len, (float*)d_out, B, N, H, T);
}
